// CoattMaxPool_8065948582127
// MI455X (gfx1250) — matrix operand order repaired and checked by reading, not compiled, not run on hardware
//
#include <hip/hip_runtime.h>

// CDNA5 / gfx1250: wave32, WMMA 16x16x32 bf16 with f32 accumulate.
typedef __bf16 bf16;
typedef __attribute__((ext_vector_type(16))) __bf16 v16bf;
typedef __attribute__((ext_vector_type(8)))  float  v8f;
typedef unsigned long long u64t;
typedef unsigned int v4u __attribute__((ext_vector_type(4)));
typedef int          v4i __attribute__((ext_vector_type(4)));
typedef int          v8i __attribute__((ext_vector_type(8)));

#define Bb 128
#define Ts 256
#define Dd 300
#define Hh 300
#define BT 32768            // Bb*Ts
#define G4 1200             // 4*Hh

// ---- Tensor Data Mover availability (probe-verified builtins; arity differs
// ---- between ROCm 7.2 (5 args) and amdgpu-toolchain/therock (6 args)).
#if defined(__HIP_DEVICE_COMPILE__) && \
    __has_builtin(__builtin_amdgcn_tensor_load_to_lds) && \
    __has_builtin(__builtin_amdgcn_s_wait_tensorcnt)
#define USE_TDM 1
#if __has_include(<hip/amd_detail/amd_gfx1250_TDM.h>)
#define TDM_CALL(g0, g1, za, zb) \
  __builtin_amdgcn_tensor_load_to_lds(g0, g1, za, zb, (v8i)0, 0)
#else
#define TDM_CALL(g0, g1, za, zb) \
  __builtin_amdgcn_tensor_load_to_lds(g0, g1, za, zb, 0)
#endif
#else
#define USE_TDM 0
#endif

#if USE_TDM
// Issue one TDM 2D tile load: 64 rows x 64 bf16, row stride = ld elements,
// into LDS at lds_byte_addr with +16B padding per 128B row
// (pad_interval=32 DWORDs -> code 4, pad_amount=4 DWORDs -> code 3),
// matching the As/Ws row pitch of (64+8) bf16 = 144B.
__device__ __forceinline__ void tdm_load_tile64(const bf16* gsrc,
                                                unsigned lds_byte_addr,
                                                int ld)
{
  unsigned long long ga = (unsigned long long)gsrc;
  v4u g0;
  g0[0] = 1u;                                            // count=1 (valid D#)
  g0[1] = lds_byte_addr;                                 // lds_addr [63:32]
  g0[2] = (unsigned)(ga & 0xffffffffu);                  // global_addr lo
  g0[3] = (unsigned)((ga >> 32) & 0x01ffffffu)           // global_addr[56:32]
        | 0x80000000u;                                   // type=2 ("image")
  v8i g1 = (v8i)0;
  g1[0] = (int)((1u << 16)        // data_size = 2 bytes
              | (1u << 20)        // pad_enable
              | (4u << 22)        // pad_interval: 32 DWORDs
              | (3u << 25));      // pad_amount:   4 DWORDs
  g1[1] = (int)(64u << 16);       // tensor_dim0 = 64  (bits [79:48])
  g1[2] = (int)(64u << 16);       // tensor_dim1 = 64  (bits [111:80])
  g1[3] = (int)(64u << 16);       // tile_dim0  = 64   (bits [127:112])
  g1[4] = (int)(64u);             // tile_dim1  = 64   (bits [143:128])
  g1[5] = (int)(unsigned)ld;      // tensor_dim0_stride (bits [207:160])
  g1[6] = 0;
  g1[7] = 0;
  TDM_CALL(g0, g1, (v4i)0, (v4i)0);
}
#endif

// =====================================================================
// Generic tiled WMMA GEMM:
//   C[z][m][n] = act( sum_k A[z][m][k] * W(k,n) + bias0[n] + bias1[n] + addC[z][m][n] )
// A: bf16 row-major [M,K] (lda), z-stride zAs elements.
// W: bf16; transB=1 -> W is [N,K] row-major (weights), element (n,k) at n*ldw+k
//          transB=0 -> W is [K,N] row-major,            element (n,k) at k*ldw+n
// addC: optional bf16 addend (row stride ldadd, z-stride zAdds).
// Output to fp32 (Cf) and/or bf16 (Cb), ldc / zCs.
//
// Tiling: BM=64 x BN=64 x BK=64. 256 threads = 8 waves, 4(M) x 2(N);
// each wave owns a 16(M) x 32(N) strip -> 2 K-substeps x 2 N-subtiles
// = 4 v_wmma per barrier pair, A-fragment shared across the N pair.
// Full interior tiles are staged via the Tensor Data Mover (TENSORcnt);
// edge tiles / transposed-W use guarded 8-byte vector loads.
//
// REQUIREMENT (satisfied by all call sites): K % 4 == 0, lda/ldw % 4 == 0,
// and all element offsets 4-aligned.
// =====================================================================
__global__ __launch_bounds__(256) void k_gemm(
    const bf16* __restrict__ A, long zAs, int lda,
    const bf16* __restrict__ W, long zWs, int ldw, int transB,
    const float* __restrict__ bias0, const float* __restrict__ bias1,
    const bf16* __restrict__ addC, long zAdds, int ldadd,
    float* __restrict__ Cf, bf16* __restrict__ Cb, long zCs, int ldc,
    int M, int N, int K, int relu)
{
  constexpr int BM = 64, BN = 64, BK = 64;
  __shared__ __align__(16) bf16 As[BM][BK + 8];
  __shared__ __align__(16) bf16 Ws[BN][BK + 8];

  const int z = blockIdx.z;
  A += (long)z * zAs;
  W += (long)z * zWs;
  if (addC) addC += (long)z * zAdds;
  const long coff = (long)z * zCs;

  const int tid  = threadIdx.x;
  const int lane = tid & 31;
  const int wave = tid >> 5;
  const int wm = wave & 3;         // 0..3  (M sub-tile)
  const int wn = wave >> 2;        // 0..1  (32-wide N half)
  const int m0 = blockIdx.y * BM;
  const int n0 = blockIdx.x * BN;

  v8f acc0 = {}, acc1 = {};
  const int nK = (K + BK - 1) / BK;

  const int mrow  = wm * 16 + (lane & 15);
  const int nrow0 = wn * 32 + (lane & 15);
  const int kh    = lane >> 4;

  for (int kt = 0; kt < nK; ++kt) {
    const int k0 = kt * BK;
    const bool fullK = (k0 + BK <= K);
    const bool fullA = fullK && (m0 + BM <= M);
    const bool fullW = fullK && (n0 + BN <= N) && transB;

    // ---- stage A tile (BM x BK) ----
#if USE_TDM
    if (fullA) {
      if (wave == 0)
        tdm_load_tile64(A + (long)m0 * lda + k0,
                        (unsigned)(unsigned long long)&As[0][0], lda);
    } else
#endif
    {
#pragma unroll
      for (int it = 0; it < 4; ++it) {
        int i  = tid + it * 256;           // 0..1023
        int r  = i >> 4;                   // row 0..63
        int cc = (i & 15) << 2;            // col 0..60 (x4)
        int gm = m0 + r, gk = k0 + cc;
        u64t v = 0;
        if (gm < M && gk < K) v = *(const u64t*)(A + (long)gm * lda + gk);
        *(u64t*)&As[r][cc] = v;
      }
    }
    // ---- stage W tile (BN x BK) ----
#if USE_TDM
    if (fullW) {
      if (wave == 1)
        tdm_load_tile64(W + (long)n0 * ldw + k0,
                        (unsigned)(unsigned long long)&Ws[0][0], ldw);
    } else
#endif
    if (transB) {
#pragma unroll
      for (int it = 0; it < 4; ++it) {
        int i  = tid + it * 256;
        int r  = i >> 4;                 // n index 0..63
        int cc = (i & 15) << 2;          // k offset
        int gn = n0 + r, gk = k0 + cc;
        u64t v = 0;
        if ((!fullW) && gn < N && gk < K)
          v = *(const u64t*)(W + (long)gn * ldw + gk);
        if (!fullW) *(u64t*)&Ws[r][cc] = v;
      }
    } else {
#pragma unroll
      for (int it = 0; it < 4; ++it) {
        int i  = tid + it * 256;
        int r  = i >> 4;                 // k index 0..63
        int cc = (i & 15) << 2;          // n offset (4 consecutive n, contiguous global)
        int gk = k0 + r, gn = n0 + cc;
        u64t v = 0;
        if (gk < K && gn < N) v = *(const u64t*)(W + (long)gk * ldw + gn);
        bf16 tmp[4];
        *(u64t*)tmp = v;
        Ws[cc + 0][r] = tmp[0];
        Ws[cc + 1][r] = tmp[1];
        Ws[cc + 2][r] = tmp[2];
        Ws[cc + 3][r] = tmp[3];
      }
    }
#if USE_TDM
    __builtin_amdgcn_s_wait_tensorcnt(0);   // TDM tiles resident before barrier
#endif
    __syncthreads();

    // ---- 2 K-substeps x 2 N-subtiles = 4 WMMAs per barrier pair ----
#pragma unroll
    for (int ks = 0; ks < 2; ++ks) {
      const int kb0 = ks * 32;
      v16bf af, bw0, bw1;
#pragma unroll
      for (int v = 0; v < 8; ++v) {
        // A frag: ISA 7.12.2, 16-bit A 16x32 (K striped across VGPRs in lane)
        int ka = kb0 + (v >> 2) * 16 + kh * 8 + (v & 3) * 2;
        af[2 * v]     = As[mrow][ka];
        af[2 * v + 1] = As[mrow][ka + 1];
        // B frag: same K striping as A frag
        int kb = kb0 + (v >> 2) * 16 + kh * 8 + (v & 3) * 2;
        bw0[2 * v]     = Ws[nrow0][kb];
        bw0[2 * v + 1] = Ws[nrow0][kb + 1];
        bw1[2 * v]     = Ws[nrow0 + 16][kb];
        bw1[2 * v + 1] = Ws[nrow0 + 16][kb + 1];
      }
      acc0 = __builtin_amdgcn_wmma_f32_16x16x32_bf16(false, af, false, bw0,
                                                     (short)0, acc0, false, false);
      acc1 = __builtin_amdgcn_wmma_f32_16x16x32_bf16(false, af, false, bw1,
                                                     (short)0, acc1, false, false);
    }
    __syncthreads();
  }

  // ---- epilogue (C layout: VGPR r -> M=r+8*(lane/16), N=lane%16) ----
#pragma unroll
  for (int j = 0; j < 2; ++j) {
    const v8f* accp = j ? &acc1 : &acc0;
    int nc = n0 + wn * 32 + j * 16 + (lane & 15);
    if (nc < N) {
      float bv = 0.0f;
      if (bias0) bv += bias0[nc];
      if (bias1) bv += bias1[nc];
#pragma unroll
      for (int r = 0; r < 8; ++r) {
        int mr = m0 + wm * 16 + r + 8 * kh;
        if (mr < M) {
          float val = (*accp)[r] + bv;
          if (addC) val += (float)addC[(long)mr * ldadd + nc];
          if (relu) val = val > 0.0f ? val : 0.0f;
          long o = coff + (long)mr * ldc + nc;
          if (Cf) Cf[o] = val;
          if (Cb) Cb[o] = (bf16)val;
        }
      }
    }
  }
}

// ---------------- embedding gather (+ per-sample reversed copy) -------------
__global__ __launch_bounds__(128) void k_embed(const int* __restrict__ s,
    const int* __restrict__ lens, const float* __restrict__ emb,
    bf16* __restrict__ p, bf16* __restrict__ pr)
{
  int bt = blockIdx.x, b = bt >> 8, t = bt & 255;
  int L = lens[b];
  int tr = (t < L) ? (L - 1 - t) : t;
  const float* e0 = emb + (long)s[bt] * Dd;
  const float* e1 = emb + (long)s[b * Ts + tr] * Dd;
  bf16* pd = p  + (long)bt * Dd;
  bf16* pq = pr + (long)bt * Dd;
  for (int d = threadIdx.x; d < Dd; d += 128) {
    pd[d] = (bf16)e0[d];
    pq[d] = (bf16)e1[d];
  }
}

// ---------------- reversed bf16 copy [B,T,300] ------------------------------
__global__ __launch_bounds__(256) void k_revbf(const bf16* __restrict__ src,
    const int* __restrict__ lens, bf16* __restrict__ dst)
{
  long idx = (long)blockIdx.x * 256 + threadIdx.x;
  if (idx >= (long)BT * Dd) return;
  int d = (int)(idx % Dd);
  long bt = idx / Dd;
  int b = (int)(bt >> 8), t = (int)(bt & 255);
  int L = lens[b];
  int tr = (t < L) ? (L - 1 - t) : t;
  dst[idx] = src[((long)(b * Ts + tr)) * Dd + d];
}

// ---------------- LSTM pointwise: 4 jobs {s1f,s2f,s1b,s2b} via blockIdx.z ---
__global__ __launch_bounds__(256) void k_lstm_point(
    const float* __restrict__ gates,                 // [4][128][1200]
    float* __restrict__ h, float* __restrict__ c,    // [4][128][300]
    bf16* __restrict__ hb,                           // [4][128][300]
    const int* __restrict__ l1, const int* __restrict__ l2,
    float* __restrict__ u1, float* __restrict__ u2,  // [B][T][600]
    bf16* __restrict__ u1b, bf16* __restrict__ u2b,  // nullable
    int t)
{
  int job = blockIdx.z;          // 0:s1 fwd, 1:s2 fwd, 2:s1 bwd, 3:s2 bwd
  int seq = job & 1, dir = job >> 1;
  int idx = blockIdx.x * 256 + threadIdx.x;
  if (idx >= Bb * Hh) return;
  int b = idx / Hh, hh = idx % Hh;
  int L = (seq ? l2 : l1)[b];

  const float* g = gates + (long)job * Bb * G4 + (long)b * G4;
  float gi = g[hh], gf = g[Hh + hh], gg = g[2 * Hh + hh], go = g[3 * Hh + hh];

  long so = (long)job * Bb * Hh + idx;
  float cold = c[so], hold = h[so];
  float si = 1.0f / (1.0f + __expf(-gi));
  float sf = 1.0f / (1.0f + __expf(-gf));
  float sg = tanhf(gg);
  float sp = 1.0f / (1.0f + __expf(-go));
  float cn = sf * cold + si * sg;
  float hn = sp * tanhf(cn);

  bool m = t < L;
  float hnew = m ? hn : hold;
  float cnew = m ? cn : cold;
  h[so] = hnew; c[so] = cnew; hb[so] = (bf16)hnew;

  int tpos = dir ? ((t < L) ? (L - 1 - t) : t) : t;  // un-reverse bwd output
  float y = m ? hn : 0.0f;
  long uo = ((long)b * Ts + tpos) * (2 * Hh) + dir * Hh + hh;
  (seq ? u2 : u1)[uo] = y;
  bf16* ub = seq ? u2b : u1b;
  if (ub) ub[uo] = (bf16)y;
}

// ---------------- masked softmax over t2 (a2), bf16 out ---------------------
__global__ __launch_bounds__(256) void k_sm2(const float* __restrict__ S,
    const int* __restrict__ l1, const int* __restrict__ l2, bf16* __restrict__ a)
{
  __shared__ float red[256];
  int b = blockIdx.y, i = blockIdx.x, j = threadIdx.x;
  int L1 = l1[b], L2 = l2[b];
  long base = ((long)b * Ts + i) * Ts;
  float x = (j < L2) ? S[base + j] : -1e9f;
  red[j] = x; __syncthreads();
  for (int s = 128; s > 0; s >>= 1) { if (j < s) red[j] = fmaxf(red[j], red[j + s]); __syncthreads(); }
  float mx = red[0]; __syncthreads();
  float e = __expf(x - mx);
  red[j] = e; __syncthreads();
  for (int s = 128; s > 0; s >>= 1) { if (j < s) red[j] += red[j + s]; __syncthreads(); }
  float r = e / red[0];
  if (i >= L1) r = 0.0f;
  a[base + j] = (bf16)r;
}

// ------ masked softmax over t1 (a1), written TRANSPOSED: a1t[b][j][i] -------
__global__ __launch_bounds__(256) void k_sm1(const float* __restrict__ S,
    const int* __restrict__ l1, const int* __restrict__ l2, bf16* __restrict__ at)
{
  __shared__ float red[256];
  int b = blockIdx.y, j = blockIdx.x, i = threadIdx.x;
  int L1 = l1[b], L2 = l2[b];
  float x = (i < L1) ? S[((long)b * Ts + i) * Ts + j] : -1e9f;
  red[i] = x; __syncthreads();
  for (int s = 128; s > 0; s >>= 1) { if (i < s) red[i] = fmaxf(red[i], red[i + s]); __syncthreads(); }
  float mx = red[0]; __syncthreads();
  float e = __expf(x - mx);
  red[i] = e; __syncthreads();
  for (int s = 128; s > 0; s >>= 1) { if (i < s) red[i] += red[i + s]; __syncthreads(); }
  float r = e / red[0];
  if (j >= L2) r = 0.0f;
  at[((long)b * Ts + j) * Ts + i] = (bf16)r;
}

// ---------------- c = [u, A, u-A, u*A] in bf16 ------------------------------
__global__ __launch_bounds__(256) void k_concat(const float* __restrict__ u,
    const float* __restrict__ A, bf16* __restrict__ cb)
{
  long idx = (long)blockIdx.x * 256 + threadIdx.x;
  if (idx >= (long)BT * 600) return;
  long row = idx / 600; int d = (int)(idx % 600);
  float uv = u[idx], av = A[idx];
  bf16* cr = cb + row * 2400;
  cr[d]        = (bf16)uv;
  cr[600 + d]  = (bf16)av;
  cr[1200 + d] = (bf16)(uv - av);
  cr[1800 + d] = (bf16)(uv * av);
}

// ---------------- masked max over time --------------------------------------
__global__ __launch_bounds__(256) void k_maxtime(const float* __restrict__ y,
    const int* __restrict__ lens, float* __restrict__ m)
{
  int idx = blockIdx.x * 256 + threadIdx.x;
  if (idx >= Bb * 600) return;
  int b = idx / 600, d = idx % 600;
  int L = lens[b];
  float mx = -3.4e38f;
  for (int t = 0; t < L; ++t) mx = fmaxf(mx, y[((long)b * Ts + t) * 600 + d]);
  m[idx] = mx;
}

// ---------------- feat = [m1, m2, |m1-m2|, m1*m2] bf16 ----------------------
__global__ __launch_bounds__(256) void k_feat(const float* __restrict__ m1,
    const float* __restrict__ m2, bf16* __restrict__ f)
{
  int idx = blockIdx.x * 256 + threadIdx.x;
  if (idx >= Bb * 2400) return;
  int b = idx / 2400, k = idx % 2400;
  float v;
  if (k < 600)        v = m1[b * 600 + k];
  else if (k < 1200)  v = m2[b * 600 + k - 600];
  else if (k < 1800)  v = fabsf(m1[b * 600 + k - 1200] - m2[b * 600 + k - 1200]);
  else                v = m1[b * 600 + k - 1800] * m2[b * 600 + k - 1800];
  f[idx] = (bf16)v;
}

// ---------------- final 300 -> 3 head ---------------------------------------
__global__ __launch_bounds__(128) void k_final(const float* __restrict__ h,
    const float* __restrict__ smW, const float* __restrict__ smB,
    float* __restrict__ out)
{
  __shared__ float red[128];
  int b = blockIdx.x, tid = threadIdx.x;
  for (int cls = 0; cls < 3; ++cls) {
    float p = 0.0f;
    for (int d = tid; d < 300; d += 128) p += h[b * 300 + d] * smW[cls * 300 + d];
    red[tid] = p; __syncthreads();
    for (int s = 64; s > 0; s >>= 1) { if (tid < s) red[tid] += red[tid + s]; __syncthreads(); }
    if (tid == 0) out[b * 3 + cls] = red[0] + smB[cls];
    __syncthreads();
  }
}

// ---------------- utility ----------------------------------------------------
__global__ __launch_bounds__(256) void k_cvt(const float* __restrict__ x,
    bf16* __restrict__ y, long n)
{
  long i = (long)blockIdx.x * 256 + threadIdx.x;
  if (i < n) y[i] = (bf16)x[i];
}
__global__ __launch_bounds__(256) void k_zero(float* __restrict__ x, long n)
{
  long i = (long)blockIdx.x * 256 + threadIdx.x;
  if (i < n) x[i] = 0.0f;
}

// =====================================================================
extern "C" void kernel_launch(void* const* d_in, const int* in_sizes, int n_in,
                              void* d_out, int out_size, void* d_ws, size_t ws_size,
                              hipStream_t stream)
{
  (void)in_sizes; (void)n_in; (void)out_size; (void)ws_size;
  const int*   s1   = (const int*)  d_in[0];
  const int*   l1i  = (const int*)  d_in[1];
  const int*   s2   = (const int*)  d_in[2];
  const int*   l2i  = (const int*)  d_in[3];
  const float* emb  = (const float*)d_in[4];
  const float* w1f_ih = (const float*)d_in[5];  const float* w1f_hh = (const float*)d_in[6];
  const float* b1f_ih = (const float*)d_in[7];  const float* b1f_hh = (const float*)d_in[8];
  const float* w1b_ih = (const float*)d_in[9];  const float* w1b_hh = (const float*)d_in[10];
  const float* b1b_ih = (const float*)d_in[11]; const float* b1b_hh = (const float*)d_in[12];
  const float* w2f_ih = (const float*)d_in[13]; const float* w2f_hh = (const float*)d_in[14];
  const float* b2f_ih = (const float*)d_in[15]; const float* b2f_hh = (const float*)d_in[16];
  const float* w2b_ih = (const float*)d_in[17]; const float* w2b_hh = (const float*)d_in[18];
  const float* b2b_ih = (const float*)d_in[19]; const float* b2b_hh = (const float*)d_in[20];
  const float* projW  = (const float*)d_in[21]; const float* projB  = (const float*)d_in[22];
  const float* mlpW   = (const float*)d_in[23]; const float* mlpB   = (const float*)d_in[24];
  const float* smW    = (const float*)d_in[25]; const float* smB    = (const float*)d_in[26];
  float* out = (float*)d_out;

  // ---- workspace carve (~1.26 GB total; buffers reused where noted) ----
  size_t off = 0;
  char* wsb = (char*)d_ws;
  auto take = [&](size_t bytes) -> void* {
    void* r = wsb + off;
    off += (bytes + 255) & ~(size_t)255;
    return r;
  };
  const long WSZ = 1200L * 300;     // lstm weight elems
  bf16* w1f_ih_b = (bf16*)take(WSZ * 2); bf16* w1f_hh_b = (bf16*)take(WSZ * 2);
  bf16* w1b_ih_b = (bf16*)take(WSZ * 2); bf16* w1b_hh_b = (bf16*)take(WSZ * 2);
  bf16* w2f_ih_b = (bf16*)take(WSZ * 2); bf16* w2f_hh_b = (bf16*)take(WSZ * 2);
  bf16* w2b_ih_b = (bf16*)take(WSZ * 2); bf16* w2b_hh_b = (bf16*)take(WSZ * 2);
  bf16* projW_b  = (bf16*)take(300L * 2400 * 2);
  bf16* mlpW_b   = (bf16*)take(300L * 2400 * 2);

  const long PSLOT = (long)BT * Dd;             // 9,830,400 elems per (seq,dir)
  const long XSLOT = (long)BT * G4;             // 39,321,600 elems per job
  bf16*  p_all = (bf16*) take(4 * PSLOT * 2);   // [p1, p2, p1r, p2r]
  bf16*  Xg    = (bf16*) take(4 * XSLOT * 2);   // reused by layer 2
  float* gates = (float*)take(4L * Bb * G4 * 4);
  float* h_all = (float*)take(4L * Bb * Hh * 4);
  float* c_all = (float*)take(4L * Bb * Hh * 4);
  bf16*  hb    = (bf16*) take(4L * Bb * Hh * 2);
  float* u1    = (float*)take((long)BT * 600 * 4);
  float* u2    = (float*)take((long)BT * 600 * 4);
  bf16*  u1b   = (bf16*) take((long)BT * 600 * 2);
  bf16*  u2b   = (bf16*) take((long)BT * 600 * 2);
  float* Smat  = (float*)take((long)Bb * Ts * Ts * 4);
  bf16*  a2b   = (bf16*) take((long)Bb * Ts * Ts * 2);
  bf16*  a1tb  = (bf16*) take((long)Bb * Ts * Ts * 2);
  float* A1    = (float*)take((long)BT * 600 * 4);   // later reused as y1
  float* A2    = (float*)take((long)BT * 600 * 4);   // later reused as y2
  bf16*  c1b   = (bf16*) take((long)BT * 2400 * 2);
  bf16*  c2b   = (bf16*) take((long)BT * 2400 * 2);
  bf16*  pc    = (bf16*) take(4 * PSLOT * 2);        // [p1c, p2c, p1c_r, p2c_r]
  float* m1    = (float*)take((long)Bb * 600 * 4);
  float* m2    = (float*)take((long)Bb * 600 * 4);
  bf16*  featb = (bf16*) take((long)Bb * 2400 * 2);
  float* hmlp  = (float*)take((long)Bb * 300 * 4);

  auto cvt = [&](const float* x, bf16* y, long n) {
    k_cvt<<<dim3((unsigned)((n + 255) / 256)), 256, 0, stream>>>(x, y, n);
  };
  cvt(w1f_ih, w1f_ih_b, WSZ); cvt(w1f_hh, w1f_hh_b, WSZ);
  cvt(w1b_ih, w1b_ih_b, WSZ); cvt(w1b_hh, w1b_hh_b, WSZ);
  cvt(w2f_ih, w2f_ih_b, WSZ); cvt(w2f_hh, w2f_hh_b, WSZ);
  cvt(w2b_ih, w2b_ih_b, WSZ); cvt(w2b_hh, w2b_hh_b, WSZ);
  cvt(projW, projW_b, 300L * 2400); cvt(mlpW, mlpW_b, 300L * 2400);

  auto gemm = [&](const bf16* A, long zAs, int lda,
                  const bf16* W, long zWs, int ldw, int transB,
                  const float* b0, const float* b1,
                  const bf16* addC, long zAdds, int ldadd,
                  float* Cf, bf16* Cb, long zCs, int ldc,
                  int M, int N, int K, int relu, int nz) {
    dim3 g((N + 63) / 64, (M + 63) / 64, nz);
    k_gemm<<<g, 256, 0, stream>>>(A, zAs, lda, W, zWs, ldw, transB, b0, b1,
                                  addC, zAdds, ldadd, Cf, Cb, zCs, ldc, M, N, K, relu);
  };

  // ---- 1) embeddings (+ reversed) ----
  k_embed<<<BT, 128, 0, stream>>>(s1, l1i, emb, p_all,             p_all + 2 * PSLOT);
  k_embed<<<BT, 128, 0, stream>>>(s2, l2i, emb, p_all + PSLOT,     p_all + 3 * PSLOT);

  // ---- 2) layer-1 input projections: Xg = p @ Wih^T + bih + bhh (bf16) ----
  gemm(p_all,             PSLOT, Dd, w1f_ih_b, 0, Dd, 1, b1f_ih, b1f_hh,
       nullptr, 0, 0, nullptr, Xg,             XSLOT, G4, BT, G4, Dd, 0, 2);
  gemm(p_all + 2 * PSLOT, PSLOT, Dd, w1b_ih_b, 0, Dd, 1, b1b_ih, b1b_hh,
       nullptr, 0, 0, nullptr, Xg + 2 * XSLOT, XSLOT, G4, BT, G4, Dd, 0, 2);

  // ---- recurrent layer runner ----
  auto run_layer = [&](bf16* XgL, const bf16* whf, const bf16* whb,
                       float* uo1, float* uo2, bf16* ub1, bf16* ub2) {
    const long hn = 4L * Bb * Hh;
    k_zero<<<dim3((unsigned)((hn + 255) / 256)), 256, 0, stream>>>(h_all, hn);
    k_zero<<<dim3((unsigned)((hn + 255) / 256)), 256, 0, stream>>>(c_all, hn);
    k_zero<<<dim3((unsigned)((hn / 2 + 255) / 256)), 256, 0, stream>>>((float*)hb, hn / 2);
    for (int t = 0; t < Ts; ++t) {
      // fwd jobs (z = s1f, s2f)
      gemm(hb,                (long)Bb * Hh, Hh, whf, 0, Hh, 1, nullptr, nullptr,
           XgL + (long)t * G4,              XSLOT, Ts * G4,
           gates,                nullptr, (long)Bb * G4, G4, Bb, G4, Hh, 0, 2);
      // bwd jobs (z = s1b, s2b)
      gemm(hb + 2L * Bb * Hh, (long)Bb * Hh, Hh, whb, 0, Hh, 1, nullptr, nullptr,
           XgL + 2 * XSLOT + (long)t * G4,  XSLOT, Ts * G4,
           gates + 2L * Bb * G4, nullptr, (long)Bb * G4, G4, Bb, G4, Hh, 0, 2);
      k_lstm_point<<<dim3(150, 1, 4), 256, 0, stream>>>(
          gates, h_all, c_all, hb, l1i, l2i, uo1, uo2, ub1, ub2, t);
    }
  };

  // ---- 3) BiLSTM layer 1 -> u1, u2 (fp32 + bf16) ----
  run_layer(Xg, w1f_hh_b, w1b_hh_b, u1, u2, u1b, u2b);

  // ---- 4) attention ----
  gemm(u1b, (long)Ts * 600, 600, u2b, (long)Ts * 600, 600, 1, nullptr, nullptr,
       nullptr, 0, 0, Smat, nullptr, (long)Ts * Ts, Ts, Ts, Ts, 600, 0, Bb);
  k_sm2<<<dim3(Ts, Bb), 256, 0, stream>>>(Smat, l1i, l2i, a2b);
  k_sm1<<<dim3(Ts, Bb), 256, 0, stream>>>(Smat, l1i, l2i, a1tb);
  gemm(a2b,  (long)Ts * Ts, Ts, u2b, (long)Ts * 600, 600, 0, nullptr, nullptr,
       nullptr, 0, 0, A1, nullptr, (long)Ts * 600, 600, Ts, 600, Ts, 0, Bb);
  gemm(a1tb, (long)Ts * Ts, Ts, u1b, (long)Ts * 600, 600, 0, nullptr, nullptr,
       nullptr, 0, 0, A2, nullptr, (long)Ts * 600, 600, Ts, 600, Ts, 0, Bb);

  // ---- 5) concat + projection (relu) -> pc (bf16) ----
  {
    unsigned nb = (unsigned)(((long)BT * 600 + 255) / 256);
    k_concat<<<nb, 256, 0, stream>>>(u1, A1, c1b);
    k_concat<<<nb, 256, 0, stream>>>(u2, A2, c2b);
  }
  gemm(c1b, 0, 2400, projW_b, 0, 2400, 1, projB, nullptr,
       nullptr, 0, 0, nullptr, pc,         0, Dd, BT, Dd, 2400, 1, 1);
  gemm(c2b, 0, 2400, projW_b, 0, 2400, 1, projB, nullptr,
       nullptr, 0, 0, nullptr, pc + PSLOT, 0, Dd, BT, Dd, 2400, 1, 1);
  {
    unsigned nb = (unsigned)(((long)BT * Dd + 255) / 256);
    k_revbf<<<nb, 256, 0, stream>>>(pc,         l1i, pc + 2 * PSLOT);
    k_revbf<<<nb, 256, 0, stream>>>(pc + PSLOT, l2i, pc + 3 * PSLOT);
  }

  // ---- 6) layer-2 input projections (reuse Xg) ----
  gemm(pc,             PSLOT, Dd, w2f_ih_b, 0, Dd, 1, b2f_ih, b2f_hh,
       nullptr, 0, 0, nullptr, Xg,             XSLOT, G4, BT, G4, Dd, 0, 2);
  gemm(pc + 2 * PSLOT, PSLOT, Dd, w2b_ih_b, 0, Dd, 1, b2b_ih, b2b_hh,
       nullptr, 0, 0, nullptr, Xg + 2 * XSLOT, XSLOT, G4, BT, G4, Dd, 0, 2);

  // ---- 7) BiLSTM layer 2 -> y1, y2 (reuse A1/A2 as y buffers) ----
  run_layer(Xg, w2f_hh_b, w2b_hh_b, A1, A2, nullptr, nullptr);

  // ---- 8) masked time-max, features, MLP, head ----
  k_maxtime<<<300, 256, 0, stream>>>(A1, l1i, m1);
  k_maxtime<<<300, 256, 0, stream>>>(A2, l2i, m2);
  k_feat<<<1200, 256, 0, stream>>>(m1, m2, featb);
  gemm(featb, 0, 2400, mlpW_b, 0, 2400, 1, mlpB, nullptr,
       nullptr, 0, 0, hmlp, nullptr, 0, 300, Bb, 300, 2400, 1, 1);
  k_final<<<Bb, 128, 0, stream>>>(hmlp, smW, smB, out);
}
